// DemLocGraphEncoder_13211319402658
// MI455X (gfx1250) — matrix operand order repaired and checked by reading, not compiled, not run on hardware
//
#include <hip/hip_runtime.h>

// ---------------------------------------------------------------------------
// Types for CDNA5 WMMA (wave32)
// ---------------------------------------------------------------------------
typedef __attribute__((ext_vector_type(16))) __bf16 v16bf;
typedef __attribute__((ext_vector_type(8)))  __bf16 v8bf;
typedef __attribute__((ext_vector_type(8)))  float  v8f;
typedef __attribute__((ext_vector_type(4)))  unsigned int u32x4;
typedef __attribute__((ext_vector_type(8)))  int i32x8;
typedef __attribute__((ext_vector_type(4)))  int i32x4;

#define BM 128   // block tile M
#define BN 64    // block tile N
#define BK 32    // K per WMMA step (bf16 K=32)
#define LDT 40   // LDS row stride in bf16 elems (32 data + 8 pad = 80 bytes)

#if __has_builtin(__builtin_amdgcn_tensor_load_to_lds) && \
    __has_builtin(__builtin_amdgcn_s_wait_tensorcnt)
#define TDM_OK 1
#else
#define TDM_OK 0
#endif

__device__ __forceinline__ unsigned lds_off(const void* p) {
  return (unsigned)(unsigned long long)(size_t)p;   // LDS aperture: addr[31:0] = LDS offset
}

#if TDM_OK
// ---------------------------------------------------------------------------
// Issue one TDM 2D tile load (bf16 elements) global -> LDS, with LDS padding
// of 16B after every 64B so rows land at an 80B (LDT) stride.
// Descriptor bitfields per CDNA5 ISA section 8.3 / 8.4.
// ---------------------------------------------------------------------------
__device__ __forceinline__ void tdm_load_tile_bf16(
    unsigned lds_byte_off, const __bf16* gtile,
    unsigned tensor_d0, unsigned tensor_d1,
    unsigned long long row_stride_elems,
    unsigned tile_d0, unsigned tile_d1)
{
  unsigned long long ga = (unsigned long long)(size_t)gtile;
  u32x4 g0;
  g0[0] = 1u;                                   // count=1, user descriptor
  g0[1] = lds_byte_off;                         // lds_addr
  g0[2] = (unsigned)ga;                         // global_addr[31:0]
  g0[3] = (unsigned)((ga >> 32) & 0x01FFFFFFu)  // global_addr[56:32]
        | (2u << 30);                           // type = 2 ("image")
  i32x8 g1;
  g1[0] = (int)((1u << 16)                      // data_size = 1 -> 2 bytes
              | (1u << 20)                      // pad_enable
              | (3u << 22)                      // pad_interval: 16 DWORDs (64B)
              | (3u << 25));                    // pad_amount:    4 DWORDs (16B)
  g1[1] = (int)((tensor_d0 & 0xFFFFu) << 16);                         // dim0[15:0]
  g1[2] = (int)(((tensor_d0 >> 16) & 0xFFFFu) |
                ((tensor_d1 & 0xFFFFu) << 16));                       // dim0[31:16], dim1[15:0]
  g1[3] = (int)(((tensor_d1 >> 16) & 0xFFFFu) |
                ((tile_d0 & 0xFFFFu) << 16));                         // dim1[31:16], tile_dim0
  g1[4] = (int)(tile_d1 & 0xFFFFu);                                   // tile_dim1 (tile_dim2=0)
  g1[5] = (int)(unsigned)(row_stride_elems & 0xFFFFFFFFull);          // dim0_stride[31:0]
  g1[6] = (int)(unsigned)((row_stride_elems >> 32) & 0xFFFFull);      // dim0_stride[47:32]
  g1[7] = 0;
  i32x4 z4 = {0, 0, 0, 0};
  i32x8 z8 = {0, 0, 0, 0, 0, 0, 0, 0};
  // 6-arg form (clang-23 / therock headers): groups 0..3, extra group, cpol
  __builtin_amdgcn_tensor_load_to_lds(g0, g1, z4, z4, z8, 0);
}
#endif

// Cooperative fallback tile copies (bf16 global -> padded LDS), 256 threads
__device__ __forceinline__ void stage_b_copy(const __bf16* gtile, size_t strideElems,
                                             int tid, __bf16* dst) {
  int r = tid >> 2;             // 0..63
  int c = (tid & 3) << 3;       // 0,8,16,24
  v8bf v = *(const v8bf*)(gtile + (size_t)r * strideElems + c);
  *(v8bf*)(dst + r * LDT + c) = v;
}
__device__ __forceinline__ void stage_a_copy_bf16(const __bf16* gtile, size_t strideElems,
                                                  int tid, __bf16* dst) {
  int r = tid >> 1;             // 0..127
  int c = (tid & 1) << 4;       // 0 or 16
  v8bf v0 = *(const v8bf*)(gtile + (size_t)r * strideElems + c);
  v8bf v1 = *(const v8bf*)(gtile + (size_t)r * strideElems + c + 8);
  v8bf* d = (v8bf*)(dst + r * LDT + c);
  d[0] = v0; d[1] = v1;
}

// ---------------------------------------------------------------------------
// WMMA GEMM: C[M,N] = act(A[M,K] * Wt^T + bias), Wt is bf16 [N][K] (pre-
// transposed weights). A is fp32 (converted while staging) or bf16 (streamed
// via TDM). Double-buffered LDS; 8 wave32s, each owning a 32x32 tile done as
// 2x2 v_wmma_f32_16x16x32_bf16 with fp32 accumulators.
// ---------------------------------------------------------------------------
template <bool A_BF16, bool OUT_BF16>
__launch_bounds__(256, 2)
__global__ void gin_gemm_wmma(const void* __restrict__ Aptr,
                              const __bf16* __restrict__ Wt,
                              const float* __restrict__ bias,
                              void* __restrict__ Cptr,
                              int M, int N, int K, int do_relu)
{
  __shared__ __align__(16) __bf16 As[2][BM * LDT];  // 2 x 10 KB
  __shared__ __align__(16) __bf16 Bs[2][BN * LDT];  // 2 x  5 KB

  const int tid  = threadIdx.x;
  const int lane = tid & 31;
  const int wave = tid >> 5;       // 0..7
  const int wm_  = wave >> 1;      // 0..3 : M quadrant
  const int wn_  = wave & 1;       // 0..1 : N half
  const int blockM = (int)blockIdx.x * BM;
  const int blockN = (int)blockIdx.y * BN;

  const float*  Af = (const float*)Aptr;
  const __bf16* Ab = (const __bf16*)Aptr;

  // fp32 A staging coordinates
  const int ar = tid >> 1;         // 0..127
  const int ac = (tid & 1) << 4;   // 0 or 16

  const int fm = lane & 15;        // fragment row/col within 16
  const int kh = lane >> 4;        // K-half selector (ISA 7.12.2)

  v8f acc[2][2] = {};

  // ---------------- prologue: fill buffer 0 ----------------
#if TDM_OK
  if (wave == 0) {
    if (A_BF16)
      tdm_load_tile_bf16(lds_off(&As[0][0]), Ab + (size_t)blockM * K,
                         (unsigned)K, (unsigned)M, (unsigned long long)K, BK, BM);
    tdm_load_tile_bf16(lds_off(&Bs[0][0]), Wt + (size_t)blockN * K,
                       (unsigned)K, (unsigned)N, (unsigned long long)K, BK, BN);
    __builtin_amdgcn_s_wait_tensorcnt(0);
  }
#else
  if (A_BF16) stage_a_copy_bf16(Ab + (size_t)blockM * K, K, tid, &As[0][0]);
  stage_b_copy(Wt + (size_t)blockN * K, K, tid, &Bs[0][0]);
#endif
  if (!A_BF16) {
    const float* gA = Af + (size_t)(blockM + ar) * K + ac;
    float4 f0 = ((const float4*)gA)[0];
    float4 f1 = ((const float4*)gA)[1];
    float4 f2 = ((const float4*)gA)[2];
    float4 f3 = ((const float4*)gA)[3];
    v8bf lo, hi;
    lo[0]=(__bf16)f0.x; lo[1]=(__bf16)f0.y; lo[2]=(__bf16)f0.z; lo[3]=(__bf16)f0.w;
    lo[4]=(__bf16)f1.x; lo[5]=(__bf16)f1.y; lo[6]=(__bf16)f1.z; lo[7]=(__bf16)f1.w;
    hi[0]=(__bf16)f2.x; hi[1]=(__bf16)f2.y; hi[2]=(__bf16)f2.z; hi[3]=(__bf16)f2.w;
    hi[4]=(__bf16)f3.x; hi[5]=(__bf16)f3.y; hi[6]=(__bf16)f3.z; hi[7]=(__bf16)f3.w;
    v8bf* dst = (v8bf*)(&As[0][ar * LDT + ac]);
    dst[0] = lo; dst[1] = hi;
  }
  __syncthreads();

  // ---------------- main pipeline ----------------
  const int nsteps = K / BK;
  for (int s = 0; s < nsteps; ++s) {
    const int buf = s & 1;
    const int nxt = buf ^ 1;
    const bool has_next = (s + 1) < nsteps;
    const int k0n = (s + 1) * BK;

    float4 f0, f1, f2, f3;
    if (has_next) {
#if TDM_OK
      if (wave == 0) {
        if (A_BF16)
          tdm_load_tile_bf16(lds_off(&As[nxt][0]), Ab + (size_t)blockM * K + k0n,
                             (unsigned)K, (unsigned)M, (unsigned long long)K, BK, BM);
        tdm_load_tile_bf16(lds_off(&Bs[nxt][0]), Wt + (size_t)blockN * K + k0n,
                           (unsigned)K, (unsigned)N, (unsigned long long)K, BK, BN);
      }
#else
      if (A_BF16) stage_a_copy_bf16(Ab + (size_t)blockM * K + k0n, K, tid, &As[nxt][0]);
      stage_b_copy(Wt + (size_t)blockN * K + k0n, K, tid, &Bs[nxt][0]);
#endif
      if (!A_BF16) {
        const float* gA = Af + (size_t)(blockM + ar) * K + (k0n + ac);
        f0 = ((const float4*)gA)[0];
        f1 = ((const float4*)gA)[1];
        f2 = ((const float4*)gA)[2];
        f3 = ((const float4*)gA)[3];
      }
    }

    // ---- fragments per ISA 7.12.2 wave32 layouts ----
    v16bf afrag[2];
    #pragma unroll
    for (int tm = 0; tm < 2; ++tm) {
      const __bf16* rowp = &As[buf][(wm_ * 32 + tm * 16 + fm) * LDT];
      v8bf lo = *(const v8bf*)(rowp + kh * 8);
      v8bf hi = *(const v8bf*)(rowp + 16 + kh * 8);
      #pragma unroll
      for (int i = 0; i < 8; ++i) { afrag[tm][i] = lo[i]; afrag[tm][i + 8] = hi[i]; }
    }
    v16bf bfrag[2];
    #pragma unroll
    for (int tn = 0; tn < 2; ++tn) {
      const __bf16* rowp = &Bs[buf][(wn_ * 32 + tn * 16 + fm) * LDT + kh * 8];
      v8bf lo = *(const v8bf*)(rowp);
      v8bf hi = *(const v8bf*)(rowp + 16);
      #pragma unroll
      for (int i = 0; i < 8; ++i) { bfrag[tn][i] = lo[i]; bfrag[tn][i + 8] = hi[i]; }
    }

    #pragma unroll
    for (int tm = 0; tm < 2; ++tm)
      #pragma unroll
      for (int tn = 0; tn < 2; ++tn)
        acc[tm][tn] = __builtin_amdgcn_wmma_f32_16x16x32_bf16(
            false, afrag[tm], false, bfrag[tn],
            (short)0, acc[tm][tn], false, false);

    if (has_next) {
      if (!A_BF16) {
        v8bf lo, hi;
        lo[0]=(__bf16)f0.x; lo[1]=(__bf16)f0.y; lo[2]=(__bf16)f0.z; lo[3]=(__bf16)f0.w;
        lo[4]=(__bf16)f1.x; lo[5]=(__bf16)f1.y; lo[6]=(__bf16)f1.z; lo[7]=(__bf16)f1.w;
        hi[0]=(__bf16)f2.x; hi[1]=(__bf16)f2.y; hi[2]=(__bf16)f2.z; hi[3]=(__bf16)f2.w;
        hi[4]=(__bf16)f3.x; hi[5]=(__bf16)f3.y; hi[6]=(__bf16)f3.z; hi[7]=(__bf16)f3.w;
        v8bf* dst = (v8bf*)(&As[nxt][ar * LDT + ac]);
        dst[0] = lo; dst[1] = hi;
      }
#if TDM_OK
      if (wave == 0) __builtin_amdgcn_s_wait_tensorcnt(0);
#endif
      __syncthreads();
    }
  }

  // ---- epilogue: bias + optional ReLU; C/D layout M=v+(lane>=16)*8, N=lane&15 ----
  const int rsel = kh * 8;
  #pragma unroll
  for (int tm = 0; tm < 2; ++tm) {
    #pragma unroll
    for (int tn = 0; tn < 2; ++tn) {
      const int c = blockN + wn_ * 32 + tn * 16 + fm;
      const float bv = bias[c];
      #pragma unroll
      for (int v = 0; v < 8; ++v) {
        const int r = blockM + wm_ * 32 + tm * 16 + rsel + v;
        float val = acc[tm][tn][v] + bv;
        if (do_relu) val = fmaxf(val, 0.0f);
        if (OUT_BF16) ((__bf16*)Cptr)[(size_t)r * N + c] = (__bf16)val;
        else          ((float*)Cptr)[(size_t)r * N + c]  = val;
      }
    }
  }
}

// ---------------------------------------------------------------------------
// Weight prep: Wt[n][k] = bf16(W[k][n])   (32x32 LDS tile transpose)
// ---------------------------------------------------------------------------
__global__ void transpose_w_bf16(const float* __restrict__ W, __bf16* __restrict__ Wt,
                                 int K, int N)
{
  __shared__ float t[32][33];
  const int kb = (int)blockIdx.x * 32;
  const int nb = (int)blockIdx.y * 32;
  const int tx = threadIdx.x & 31;
  const int ty = threadIdx.x >> 5;          // 0..7
  #pragma unroll
  for (int i = ty; i < 32; i += 8)
    t[i][tx] = W[(size_t)(kb + i) * N + (nb + tx)];
  __syncthreads();
  #pragma unroll
  for (int i = ty; i < 32; i += 8)
    Wt[(size_t)(nb + i) * K + (kb + tx)] = (__bf16)t[tx][i];
}

// ---------------------------------------------------------------------------
// agg = x   (seed for (1+eps)*x + sum_j x_j with eps=0)
// ---------------------------------------------------------------------------
__global__ void copy_f32_kernel(const float* __restrict__ src,
                                float* __restrict__ dst, long n4)
{
  long i = (long)blockIdx.x * blockDim.x + threadIdx.x;
  if (i < n4) ((float4*)dst)[i] = ((const float4*)src)[i];
}

// ---------------------------------------------------------------------------
// agg[dst[e]] += x[src[e]]  (float4 gather + f32 global atomics; the 64 MB
// activation lives in the 192 MB L2, so gathers/atomics resolve in L2)
// ---------------------------------------------------------------------------
__global__ void edge_scatter_kernel(const float* __restrict__ x,
                                    const int* __restrict__ idx,   // [2,E]
                                    float* __restrict__ agg,
                                    int E, int d)
{
  long t = (long)blockIdx.x * blockDim.x + threadIdx.x;
  const int chunks = d >> 2;
  long total = (long)E * chunks;
  if (t >= total) return;
  int e = (int)(t / chunks);
  int f = (int)(t % chunks) << 2;
  int s = idx[e];
  int dn = idx[E + e];
  float4 v = *(const float4*)(x + (size_t)s * d + f);
  float* p = agg + (size_t)dn * d + f;
  atomicAdd(p + 0, v.x);
  atomicAdd(p + 1, v.y);
  atomicAdd(p + 2, v.z);
  atomicAdd(p + 3, v.w);
}

// ---------------------------------------------------------------------------
// z = mean + var * eps
// ---------------------------------------------------------------------------
__global__ void reparam_kernel(const float* __restrict__ mean,
                               const float* __restrict__ var,
                               const float* __restrict__ eps,
                               float* __restrict__ z, long n)
{
  long i = (long)blockIdx.x * blockDim.x + threadIdx.x;
  if (i < n) z[i] = fmaf(var[i], eps[i], mean[i]);
}

// ---------------------------------------------------------------------------
// Orchestration
// ---------------------------------------------------------------------------
extern "C" void kernel_launch(void* const* d_in, const int* in_sizes, int n_in,
                              void* d_out, int out_size, void* d_ws, size_t ws_size,
                              hipStream_t stream)
{
  (void)in_sizes; (void)n_in; (void)out_size; (void)ws_size;
  const int Nn = 8192, E = 262144, T = 256, H = 2048, O = 1024, L = 128;

  const float* x0  = (const float*)d_in[0];
  const int*   idx = (const int*)d_in[1];
  const float* w1[4] = {(const float*)d_in[2],  (const float*)d_in[6],
                        (const float*)d_in[10], (const float*)d_in[14]};
  const float* b1[4] = {(const float*)d_in[3],  (const float*)d_in[7],
                        (const float*)d_in[11], (const float*)d_in[15]};
  const float* w2[4] = {(const float*)d_in[4],  (const float*)d_in[8],
                        (const float*)d_in[12], (const float*)d_in[16]};
  const float* b2[4] = {(const float*)d_in[5],  (const float*)d_in[9],
                        (const float*)d_in[13], (const float*)d_in[17]};
  const float* wm  = (const float*)d_in[18];
  const float* bm  = (const float*)d_in[19];
  const float* wv  = (const float*)d_in[20];
  const float* bv  = (const float*)d_in[21];
  const float* eps = (const float*)d_in[22];

  // workspace carve-up
  char* ws = (char*)d_ws;
  __bf16* wt   = (__bf16*)ws;                                   //  8 MB bf16 [N][K]
  float*  agg  = (float*)(ws + (size_t)8  * 1024 * 1024);       // 64 MB f32
  __bf16* hbuf = (__bf16*)(ws + (size_t)72 * 1024 * 1024);      // 32 MB bf16
  float*  xbuf = (float*)(ws + (size_t)104 * 1024 * 1024);      // 64 MB f32

  float* outz = (float*)d_out;                 // tuple (z, mean, var) concat
  float* outm = outz + (size_t)Nn * L;
  float* outv = outm + (size_t)Nn * L;

  const int dims_in[4]  = {T, H, H, H};
  const int dims_mid[4] = {H, H, H, H};
  const int dims_out[4] = {H, H, H, O};

  const float* xin = x0;
  for (int l = 0; l < 4; ++l) {
    const int din  = dims_in[l];
    const int dmid = dims_mid[l];
    const int dout = dims_out[l];

    // agg = x + segment_sum(x[src] -> dst)
    long n4 = (long)Nn * din / 4;
    copy_f32_kernel<<<(unsigned)((n4 + 255) / 256), 256, 0, stream>>>(xin, agg, n4);
    long etot = (long)E * (din / 4);
    edge_scatter_kernel<<<(unsigned)((etot + 255) / 256), 256, 0, stream>>>(
        xin, idx, agg, E, din);

    // h = relu(agg @ w1 + b1)          (fp32 A, bf16 out)
    transpose_w_bf16<<<dim3(din / 32, dmid / 32), 256, 0, stream>>>(w1[l], wt, din, dmid);
    gin_gemm_wmma<false, true><<<dim3(Nn / BM, dmid / BN), 256, 0, stream>>>(
        agg, wt, b1[l], hbuf, Nn, dmid, din, 1);

    // x' = h @ w2 + b2  (+ inter-layer ReLU, except last)   (bf16 A via TDM, f32 out)
    transpose_w_bf16<<<dim3(dmid / 32, dout / 32), 256, 0, stream>>>(w2[l], wt, dmid, dout);
    gin_gemm_wmma<true, false><<<dim3(Nn / BM, dout / BN), 256, 0, stream>>>(
        hbuf, wt, b2[l], xbuf, Nn, dout, dmid, (l < 3) ? 1 : 0);

    xin = xbuf;
  }

  // heads: mean / var projections (K = 1024, N = 128), fp32 A, f32 out
  transpose_w_bf16<<<dim3(O / 32, L / 32), 256, 0, stream>>>(wm, wt, O, L);
  gin_gemm_wmma<false, false><<<dim3(Nn / BM, L / BN), 256, 0, stream>>>(
      xbuf, wt, bm, outm, Nn, L, O, 0);
  transpose_w_bf16<<<dim3(O / 32, L / 32), 256, 0, stream>>>(wv, wt, O, L);
  gin_gemm_wmma<false, false><<<dim3(Nn / BM, L / BN), 256, 0, stream>>>(
      xbuf, wt, bv, outv, Nn, L, O, 0);

  long nz = (long)Nn * L;
  reparam_kernel<<<(unsigned)((nz + 255) / 256), 256, 0, stream>>>(
      outm, outv, eps, outz, nz);
}
